// MultiHeadCrossAttention_20529943675563
// MI455X (gfx1250) — matrix operand order repaired and checked by reading, not compiled, not run on hardware
//
#include <hip/hip_runtime.h>
#include <hip/hip_bf16.h>

typedef __attribute__((ext_vector_type(16))) __bf16 v16bf;
typedef __attribute__((ext_vector_type(8)))  float  v8f;

#define WMMA_BF16(a, b, c) \
    __builtin_amdgcn_wmma_f32_16x16x32_bf16(false, (a), false, (b), (short)0, (c), false, false)

// ---------------- problem constants ----------------
static constexpr int B     = 8;
static constexpr int N1    = 5,  L1 = N1 * 144;   // 720 keys
static constexpr int N2    = 15, L2 = N2 * 144;   // 2160 queries
static constexpr int L1P   = 768;                 // key pad (12 tiles of 64)
static constexpr int L2P   = 2176;                // query pad (68 chunks of 32)
static constexpr int KVLD  = 736;                 // 720 padded to 23 chunks of 32
static constexpr float SCALE = 0.08838834764831845f; // 128^-0.5

// ---------------- workspace layout (bytes) ----------------
static constexpr size_t OFF_WT    = 0;                                   // 5 x 512x512 bf16 transposed
static constexpr size_t SZ_WT     = 5ull * 512 * 512 * 2;
static constexpr size_t OFF_F2TOK = OFF_WT + SZ_WT;                      // [B*L2,512] bf16
static constexpr size_t SZ_F2TOK  = (size_t)B * L2 * 512 * 2;
static constexpr size_t OFF_F1TOK = OFF_F2TOK + SZ_F2TOK;                // [B*L1,512] bf16
static constexpr size_t SZ_F1TOK  = (size_t)B * L1 * 512 * 2;
static constexpr size_t OFF_QRM   = OFF_F1TOK + SZ_F1TOK;                // [32][L2P][128] bf16
static constexpr size_t SZ_QRM    = 32ull * L2P * 128 * 2;
static constexpr size_t OFF_QT    = OFF_QRM + SZ_QRM;                    // [32][128][L2P] bf16
static constexpr size_t SZ_QT     = 32ull * 128 * L2P * 2;
static constexpr size_t OFF_KH    = OFF_QT + SZ_QT;                      // [32][L1P][128] bf16
static constexpr size_t SZ_KH     = 32ull * L1P * 128 * 2;
static constexpr size_t OFF_VT    = OFF_KH + SZ_KH;                      // [32][128][KVLD] bf16
static constexpr size_t SZ_VT     = 32ull * 128 * KVLD * 2;
static constexpr size_t OFF_O2    = OFF_VT + SZ_VT;                      // [B*L2,512] bf16
static constexpr size_t SZ_O2     = (size_t)B * L2 * 512 * 2;
static constexpr size_t OFF_O1    = OFF_O2 + SZ_O2;                      // [B*L1,512] bf16
static constexpr size_t SZ_O1     = (size_t)B * L1 * 512 * 2;
static constexpr size_t OFF_SM    = OFF_O1 + SZ_O1;                      // [32*L2] f32 row max
static constexpr size_t SZ_SM     = 32ull * L2 * 4;
static constexpr size_t OFF_SS    = OFF_SM + SZ_SM;                      // [32*L2] f32 row sum

// ---------------- small prep kernels ----------------

__global__ __launch_bounds__(256) void k_wconv(const float* __restrict__ W, __bf16* __restrict__ WT) {
    int idx = blockIdx.x * 256 + threadIdx.x;      // 512*512
    int n = idx & 511, k = idx >> 9;
    WT[(size_t)n * 512 + k] = (__bf16)W[idx];
}

__global__ __launch_bounds__(256) void k_tok(const float* __restrict__ src, __bf16* __restrict__ dst,
                                             int total) {
    int idx = blockIdx.x * 256 + threadIdx.x;
    if (idx >= total) return;
    int hw = idx % 144;
    int c  = (idx / 144) & 511;
    int bi = idx / (144 * 512);                    // b*n + i
    size_t t = (size_t)bi * 144 + hw;              // token row
    dst[t * 512 + c] = (__bf16)src[idx];
}

__global__ __launch_bounds__(256) void k_zero_qtpad(__bf16* __restrict__ QT) {
    int idx = blockIdx.x * 256 + threadIdx.x;      // 32*128*16
    int r = idx >> 4;
    int c = L2 + (idx & 15);
    QT[(size_t)r * L2P + c] = (__bf16)0.f;
}

// ---------------- fragment helpers (contiguous k = 16*(lane/16) + s, wave32) ----------------
__device__ __forceinline__ v16bf frag_a(const __bf16* p, int koff) {  // p = row base (+kb)
    v16bf a;
#pragma unroll
    for (int i = 0; i < 8; ++i) { a[i] = p[2 * koff + i]; a[i + 8] = p[2 * koff + 8 + i]; }
    return a;
}
__device__ __forceinline__ v16bf frag_b(const __bf16* p) {            // 16 contiguous bf16
    v16bf b;
#pragma unroll
    for (int i = 0; i < 16; ++i) b[i] = p[i];
    return b;
}

// ---------------- generic bf16 GEMM: C[M,512] = A[M,512] @ W + bias ----------------
// MODE 0: head-split row-major bf16   MODE 1: head-split transposed bf16
// MODE 2: both                        MODE 3: fp32 untokenize scatter
template <int MODE, int LTOK, int NIMG, int LPAD, int TLD>
__global__ __launch_bounds__(128) void k_gemm(const __bf16* __restrict__ A,
                                              const __bf16* __restrict__ WT,
                                              const float* __restrict__ bias,
                                              __bf16* __restrict__ outRM,
                                              __bf16* __restrict__ outT,
                                              float* __restrict__ outF) {
    int w = threadIdx.x >> 5, lane = threadIdx.x & 31;
    int m = lane & 15, koff = (lane >> 4) * 8, kh = (lane >> 4) * 16;
    int row0 = blockIdx.x * 64 + w * 16;
    int col0 = blockIdx.y * 64;
    const __bf16* arow = A + (size_t)(row0 + m) * 512;
    const __bf16* wrow[4];
#pragma unroll
    for (int dt = 0; dt < 4; ++dt) wrow[dt] = WT + (size_t)(col0 + dt * 16 + m) * 512 + kh;

    v8f acc[4] = {};
    // software pipeline, prefetch distance 1, fully unrolled (15 + epilogue)
    v16bf a0 = frag_a(arow, koff);
    v16bf b0[4];
#pragma unroll
    for (int dt = 0; dt < 4; ++dt) b0[dt] = frag_b(wrow[dt]);
#pragma unroll
    for (int kb = 0; kb < 480; kb += 32) {
        v16bf a1 = frag_a(arow + kb + 32, koff);
        v16bf b1[4];
#pragma unroll
        for (int dt = 0; dt < 4; ++dt) b1[dt] = frag_b(wrow[dt] + kb + 32);
#pragma unroll
        for (int dt = 0; dt < 4; ++dt) acc[dt] = WMMA_BF16(a0, b0[dt], acc[dt]);
        a0 = a1;
#pragma unroll
        for (int dt = 0; dt < 4; ++dt) b0[dt] = b1[dt];
    }
#pragma unroll
    for (int dt = 0; dt < 4; ++dt) acc[dt] = WMMA_BF16(a0, b0[dt], acc[dt]);

#pragma unroll
    for (int dt = 0; dt < 4; ++dt) {
        int col = col0 + dt * 16 + m;
        float bb = bias[col];
#pragma unroll
        for (int v = 0; v < 8; ++v) {
            int row = row0 + v + (lane >> 4) * 8;
            float val = acc[dt][v] + bb;
            int b = row / LTOK, l = row % LTOK;   // compile-time divisor
            if constexpr (MODE == 3) {
                int i = l / 144, hw = l % 144;
                outF[((size_t)(b * NIMG + i) * 512 + col) * 144 + hw] = val;
            } else {
                int h = col >> 7, d = col & 127;
                if constexpr (MODE == 0 || MODE == 2)
                    outRM[((size_t)(b * 4 + h) * LPAD + l) * 128 + d] = (__bf16)val;
                if constexpr (MODE == 1 || MODE == 2)
                    outT[((size_t)(b * 4 + h) * 128 + d) * (size_t)TLD + l] = (__bf16)val;
            }
        }
    }
}

// ---------------- attention pass 1: softmax(QK^T)V -> O2, + row stats ----------------
__global__ __launch_bounds__(128) void k_att2(const __bf16* __restrict__ Qrm,
                                              const __bf16* __restrict__ Kh,
                                              const __bf16* __restrict__ Vt,
                                              __bf16* __restrict__ O2,
                                              float* __restrict__ statM,
                                              float* __restrict__ statS) {
    extern __shared__ char smem[];
    __bf16* sQ = (__bf16*)smem;                             // [32][128] bf16
    float*  sS = (float*)(smem + 32 * 128 * 2);             // [32][KVLD] f32
    float*  sR = (float*)(smem + 32 * 128 * 2 + 32 * KVLD * 4); // [4][32] f32 reduce
    int tid = threadIdx.x;
    int bh = blockIdx.y, q0 = blockIdx.x * 32;
    int b = bh >> 2, h = bh & 3;
    const __bf16* Qb = Qrm + (size_t)bh * L2P * 128;
    const __bf16* Kb = Kh  + (size_t)bh * L1P * 128;
    const __bf16* Vb = Vt  + (size_t)bh * 128 * KVLD;

    for (int i = tid; i < 32 * 128; i += 128) {
        int r = i >> 7, c = i & 127, q = q0 + r;
        sQ[i] = (q < L2) ? Qb[(size_t)q * 128 + c] : (__bf16)0.f;
    }
    for (int i = tid; i < 32 * 16; i += 128) sS[(i >> 4) * KVLD + 720 + (i & 15)] = 0.f;
    __syncthreads();

    int w = tid >> 5, lane = tid & 31;
    int m = lane & 15, koff = (lane >> 4) * 8, kh = (lane >> 4) * 16;
    int qh = w & 1;
    const __bf16* aQ = sQ + (size_t)(qh * 16 + m) * 128;

    // ---- S = Q K^T : Q A-fragments hoisted; peeled pipeline, no conditional prefetch ----
    v16bf aF[4];
#pragma unroll
    for (int dc = 0; dc < 4; ++dc) aF[dc] = frag_a(aQ + dc * 32, koff);

    auto ldK = [&](int kt, v16bf* d) {
        const __bf16* kp = Kb + (size_t)(kt * 16 + m) * 128 + kh;
#pragma unroll
        for (int dc = 0; dc < 4; ++dc) d[dc] = frag_b(kp + dc * 32);
    };
    auto stS = [&](int kt, v8f acc) {
        float* srow = sS + (size_t)(qh * 16 + (lane >> 4) * 8) * KVLD + kt * 16 + m;
#pragma unroll
        for (int v = 0; v < 8; ++v) srow[(size_t)v * KVLD] = acc[v] * SCALE;
    };

    int kt = (w >> 1);
    v16bf bK[4];
    ldK(kt, bK);
#pragma unroll 2
    for (; kt + 2 < 45; kt += 2) {
        v16bf bN[4];
        ldK(kt + 2, bN);                      // unconditional prefetch
        v8f acc = {};
#pragma unroll
        for (int dc = 0; dc < 4; ++dc) acc = WMMA_BF16(aF[dc], bK[dc], acc);
        stS(kt, acc);
#pragma unroll
        for (int dc = 0; dc < 4; ++dc) bK[dc] = bN[dc];
    }
    {                                          // peeled final iteration
        v8f acc = {};
#pragma unroll
        for (int dc = 0; dc < 4; ++dc) acc = WMMA_BF16(aF[dc], bK[dc], acc);
        stS(kt, acc);
    }
    __syncthreads();

    // ---- split-row softmax: 4 threads per row, 180 columns each (720 = 4*180) ----
    {
        int r = tid & 31, p = tid >> 5;
        float* row = sS + (size_t)r * KVLD;
        int c0 = p * 180;
        float mx = -1e30f;
        for (int c = c0; c < c0 + 180; ++c) mx = fmaxf(mx, row[c]);
        sR[p * 32 + r] = mx;
        __syncthreads();
        float gm = fmaxf(fmaxf(sR[r], sR[32 + r]), fmaxf(sR[64 + r], sR[96 + r]));
        float s = 0.f;
        for (int c = c0; c < c0 + 180; ++c) { float e = __expf(row[c] - gm); row[c] = e; s += e; }
        __syncthreads();                       // all gm reads done before sR reuse
        sR[p * 32 + r] = s;
        __syncthreads();
        float gs = sR[r] + sR[32 + r] + sR[64 + r] + sR[96 + r];
        float inv = 1.f / gs;
        for (int c = c0; c < c0 + 180; ++c) row[c] *= inv;
        int q = q0 + r;
        if (p == 0 && q < L2) { statM[(size_t)bh * L2 + q] = gm; statS[(size_t)bh * L2 + q] = gs; }
    }
    __syncthreads();

    // ---- O2 = att @ V : kc outer, 4 dt accumulators, fully unrolled pipeline ----
    int dt0 = (w >> 1);
    const float* arow = sS + (size_t)(qh * 16 + m) * KVLD;
    auto ldA = [&](int kc) {
        v16bf a;
#pragma unroll
        for (int i = 0; i < 8; ++i) {
            a[i]     = (__bf16)arow[kc * 32 + 2 * koff + i];
            a[i + 8] = (__bf16)arow[kc * 32 + 2 * koff + 8 + i];
        }
        return a;
    };
    auto ldV = [&](int kc, v16bf* d) {
#pragma unroll
        for (int j = 0; j < 4; ++j)
            d[j] = frag_b(Vb + (size_t)((dt0 + 2 * j) * 16 + m) * KVLD + kc * 32 + kh);
    };
    v8f acc2[4] = {};
    v16bf aC = ldA(0);
    v16bf bV[4];
    ldV(0, bV);
#pragma unroll
    for (int kc = 0; kc < 22; ++kc) {
        v16bf aN = ldA(kc + 1);
        v16bf bN[4];
        ldV(kc + 1, bN);
#pragma unroll
        for (int j = 0; j < 4; ++j) acc2[j] = WMMA_BF16(aC, bV[j], acc2[j]);
        aC = aN;
#pragma unroll
        for (int j = 0; j < 4; ++j) bV[j] = bN[j];
    }
#pragma unroll
    for (int j = 0; j < 4; ++j) acc2[j] = WMMA_BF16(aC, bV[j], acc2[j]);

#pragma unroll
    for (int j = 0; j < 4; ++j) {
        int dt = dt0 + 2 * j;
#pragma unroll
        for (int v = 0; v < 8; ++v) {
            int q = q0 + qh * 16 + v + (lane >> 4) * 8;
            if (q < L2)
                O2[((size_t)b * L2 + q) * 512 + h * 128 + dt * 16 + m] = (__bf16)acc2[j][v];
        }
    }
}

// ---------------- attention pass 2: att^T @ Q -> O1 (recompute S^T, use stats) ----------------
__global__ __launch_bounds__(128) void k_att1(const __bf16* __restrict__ Kh,
                                              const __bf16* __restrict__ Qrm,
                                              const __bf16* __restrict__ QT,
                                              const float* __restrict__ statM,
                                              const float* __restrict__ statS,
                                              __bf16* __restrict__ O1) {
    __shared__ __bf16 attL[4][16 * 40];             // per-wave att staging (ld=40)
    int tid = threadIdx.x, w = tid >> 5, lane = tid & 31;
    int m = lane & 15, koff = (lane >> 4) * 8, kh = (lane >> 4) * 16;
    int bh = blockIdx.y, b = bh >> 2, h = bh & 3;
    int key0 = blockIdx.x * 64 + w * 16;
    const __bf16* Kb  = Kh  + (size_t)bh * L1P * 128;
    const __bf16* Qb  = Qrm + (size_t)bh * L2P * 128;
    const __bf16* QTb = QT  + (size_t)bh * 128 * L2P;

    v16bf ka[4];                                    // K tile A-fragments, hoisted
#pragma unroll
    for (int dc = 0; dc < 4; ++dc)
        ka[dc] = frag_a(Kb + (size_t)(key0 + m) * 128 + dc * 32, koff);

    v8f acc[8] = {};
    __bf16* aw = &attL[w][0];

    for (int qc = 0; qc < 68; ++qc) {               // 68 chunks of 32 queries (padded)
        // stream the 8 QT B-fragments early: latency hides behind S^T + exp work
        v16bf bq[8];
#pragma unroll
        for (int dt = 0; dt < 8; ++dt)
            bq[dt] = frag_b(QTb + (size_t)(dt * 16 + m) * L2P + qc * 32 + kh);

#pragma unroll
        for (int half = 0; half < 2; ++half) {
            int qb = qc * 32 + half * 16;
            v8f st = {};
#pragma unroll
            for (int dc = 0; dc < 4; ++dc)          // S^T = K . Q^T
                st = WMMA_BF16(ka[dc], frag_b(Qb + (size_t)(qb + m) * 128 + dc * 32 + kh), st);
            int q = qb + m;                         // lane column = query -> scalar stats
            bool valid = q < L2;
            int qi = valid ? q : 0;
            float mq  = statM[(size_t)bh * L2 + qi];
            float inv = 1.f / statS[(size_t)bh * L2 + qi];
#pragma unroll
            for (int v = 0; v < 8; ++v) {
                float att = valid ? __expf(st[v] * SCALE - mq) * inv : 0.f;
                aw[(v + (lane >> 4) * 8) * 40 + half * 16 + m] = (__bf16)att;
            }
        }
        v16bf a;                                    // att^T chunk as A-fragment
#pragma unroll
        for (int i = 0; i < 8; ++i) { a[i] = aw[m * 40 + 2 * koff + i]; a[i + 8] = aw[m * 40 + 2 * koff + 8 + i]; }
#pragma unroll
        for (int dt = 0; dt < 8; ++dt)              // accumulate att^T @ Q
            acc[dt] = WMMA_BF16(a, bq[dt], acc[dt]);
    }
#pragma unroll
    for (int dt = 0; dt < 8; ++dt)
#pragma unroll
        for (int v = 0; v < 8; ++v) {
            int key = key0 + v + (lane >> 4) * 8;
            if (key < L1)
                O1[((size_t)b * L1 + key) * 512 + h * 128 + dt * 16 + m] = (__bf16)acc[dt][v];
        }
}

// ---------------- host launcher ----------------
extern "C" void kernel_launch(void* const* d_in, const int* in_sizes, int n_in,
                              void* d_out, int out_size, void* d_ws, size_t ws_size,
                              hipStream_t stream) {
    const float* f1  = (const float*)d_in[0];
    const float* f2  = (const float*)d_in[1];
    const float* Wq  = (const float*)d_in[2];  const float* bq  = (const float*)d_in[3];
    const float* Wk  = (const float*)d_in[4];  const float* bk  = (const float*)d_in[5];
    const float* Wv  = (const float*)d_in[6];  const float* bv  = (const float*)d_in[7];
    const float* Woq = (const float*)d_in[8];  const float* boq = (const float*)d_in[9];
    const float* Wok = (const float*)d_in[10]; const float* bok = (const float*)d_in[11];
    char* ws = (char*)d_ws;
    __bf16* WT    = (__bf16*)(ws + OFF_WT);
    __bf16* f2tok = (__bf16*)(ws + OFF_F2TOK);
    __bf16* f1tok = (__bf16*)(ws + OFF_F1TOK);
    __bf16* Qrm   = (__bf16*)(ws + OFF_QRM);
    __bf16* QT    = (__bf16*)(ws + OFF_QT);
    __bf16* Kh    = (__bf16*)(ws + OFF_KH);
    __bf16* Vt    = (__bf16*)(ws + OFF_VT);
    __bf16* O2    = (__bf16*)(ws + OFF_O2);
    __bf16* O1    = (__bf16*)(ws + OFF_O1);
    float*  sM    = (float*)(ws + OFF_SM);
    float*  sS    = (float*)(ws + OFF_SS);
    float*  out1  = (float*)d_out;
    float*  out2  = (float*)d_out + (size_t)B * N1 * 512 * 144;

    const float* Ws[5] = {Wq, Wk, Wv, Woq, Wok};
    for (int i = 0; i < 5; ++i)
        k_wconv<<<1024, 256, 0, stream>>>(Ws[i], WT + (size_t)i * 512 * 512);
    k_tok<<<(B * N2 * 512 * 144 + 255) / 256, 256, 0, stream>>>(f2, f2tok, B * N2 * 512 * 144);
    k_tok<<<(B * N1 * 512 * 144 + 255) / 256, 256, 0, stream>>>(f1, f1tok, B * N1 * 512 * 144);
    k_zero_qtpad<<<(32 * 128 * 16) / 256, 256, 0, stream>>>(QT);

    // projections: Q (both layouts), K (row-major padded), V (transposed)
    k_gemm<2, L2, 0, L2P, L2P><<<dim3(B * L2 / 64, 8), 128, 0, stream>>>(
        f2tok, WT + 0 * 512 * 512, bq, Qrm, QT, nullptr);
    k_gemm<0, L1, 0, L1P, 1><<<dim3(B * L1 / 64, 8), 128, 0, stream>>>(
        f1tok, WT + 1 * 512 * 512, bk, Kh, nullptr, nullptr);
    k_gemm<1, L1, 0, 1, KVLD><<<dim3(B * L1 / 64, 8), 128, 0, stream>>>(
        f1tok, WT + 2 * 512 * 512, bv, nullptr, Vt, nullptr);
    // attention
    k_att2<<<dim3(68, 32), 128, 32 * 128 * 2 + 32 * KVLD * 4 + 4 * 32 * 4, stream>>>(
        Qrm, Kh, Vt, O2, sM, sS);
    k_att1<<<dim3(12, 32), 128, 0, stream>>>(Kh, Qrm, QT, sM, sS, O1);
    // output projections with fused untokenize
    k_gemm<3, L2, 15, 1, 1><<<dim3(B * L2 / 64, 8), 128, 0, stream>>>(
        O2, WT + 3 * 512 * 512, boq, nullptr, nullptr, out2);
    k_gemm<3, L1, 5, 1, 1><<<dim3(B * L1 / 64, 8), 128, 0, stream>>>(
        O1, WT + 4 * 512 * 512, bok, nullptr, nullptr, out1);
}
